// HSRnnForCausalLM_4767413699190
// MI455X (gfx1250) — matrix operand order repaired and checked by reading, not compiled, not run on hardware
//
#include <hip/hip_runtime.h>
#include <hip/hip_bf16.h>

typedef unsigned int   u32;
typedef unsigned short u16;
typedef __attribute__((ext_vector_type(4)))  u32    u32x4;
typedef __attribute__((ext_vector_type(16))) __bf16 v16bf;
typedef __attribute__((ext_vector_type(8)))  float  v8f;

union Frag { v16bf v; u32x4 q[2]; };

constexpr int Bc = 4, Sc = 1024, Ic = 512, Hc = 1024, Vc = 32000;
constexpr int NBLK = 64;   // persistent blocks for the recurrent scan (1 tile of 16 cols each)
constexpr int NW   = 8;    // waves per 256-thread block (wave32)

// ---------------- helpers ----------------

__device__ inline u16 f2bf(float f) {
  u32 u = __float_as_uint(f);
  u32 r = u + 0x7FFFu + ((u >> 16) & 1u);   // round-to-nearest-even
  return (u16)(r >> 16);
}

// One wave accumulates a 16x16 f32 tile over K in [k0,k1) using bf16 WMMA.
// A: row-major [16][lda] bf16 (callers guarantee 16 valid/zero-padded rows).
// Wt: output-major weights [N][ldb] bf16 (K contiguous) -> each lane loads B as two 16B runs.
__device__ inline v8f wave_gemm_bf16(const u16* __restrict__ Abase, size_t lda,
                                     const u16* __restrict__ Wt, int ldb, int colb,
                                     int k0, int k1, v8f acc) {
  const int lane = threadIdx.x & 31;
  const int m    = lane & 15;      // A row / B col within tile
  const int hi   = lane >> 4;
  const u16* arow = Abase + (size_t)m * lda + hi * 8;
  const u16* bcol = Wt + (size_t)(colb + m) * ldb + hi * 8;
  for (int k = k0; k < k1; k += 32) {
    Frag a, b;
    a.q[0] = *(const u32x4*)(arow + k);        // K = k+8hi .. +7
    a.q[1] = *(const u32x4*)(arow + k + 16);   // K = k+16+8hi .. +7
    b.q[0] = *(const u32x4*)(bcol + k);        // K = k+8hi .. +7
    b.q[1] = *(const u32x4*)(bcol + k + 16);   // K = k+16+8hi .. +7
    acc = __builtin_amdgcn_wmma_f32_16x16x32_bf16(false, a.v, false, b.v,
                                                  (short)0, acc, false, false);
  }
  return acc;
}

// Reduce the 8 per-wave partial 16x16 tiles; returns element (tid>>4, tid&15).
__device__ inline float block_reduce16(float (*part)[16][16], v8f acc) {
  const int lane = threadIdx.x & 31, w = threadIdx.x >> 5;
  const int hi = lane >> 4, n = lane & 15;
#pragma unroll
  for (int r = 0; r < 8; ++r) part[w][r + 8 * hi][n] = acc[r];  // C layout: M=r+8hi, N=lane&15
  __syncthreads();
  const int mm = threadIdx.x >> 4, nn = threadIdx.x & 15;
  float s = 0.f;
#pragma unroll
  for (int ww = 0; ww < NW; ++ww) s += part[ww][mm][nn];
  return s;
}

// Device-scope software barrier across the NBLK persistent blocks.
__device__ inline void grid_sync(unsigned* bar, unsigned& epoch) {
  __threadfence();
  __syncthreads();
  if (threadIdx.x == 0) {
    epoch += 1;
    const unsigned target = epoch * (unsigned)NBLK;
    __hip_atomic_fetch_add(bar, 1u, __ATOMIC_RELEASE, __HIP_MEMORY_SCOPE_AGENT);
    while (__hip_atomic_load(bar, __ATOMIC_ACQUIRE, __HIP_MEMORY_SCOPE_AGENT) < target) {
      __builtin_amdgcn_s_sleep(2);
    }
  }
  __syncthreads();
}

// ---------------- prep kernels ----------------

__global__ void k_cvt_bf16(const float* __restrict__ in, u16* __restrict__ out, size_t n) {
  size_t i = (size_t)blockIdx.x * blockDim.x + threadIdx.x;
  const size_t stride = (size_t)gridDim.x * blockDim.x;
  for (; i < n; i += stride) out[i] = f2bf(in[i]);
}

// in: [K][N] f32 row-major  ->  out: [N][K] bf16 (K contiguous). Tiled via LDS.
__global__ void k_transpose_bf16(const float* __restrict__ in, u16* __restrict__ out,
                                 int K, int N) {
  __shared__ float tile[32][33];
  const int nb = blockIdx.x * 32, kb = blockIdx.y * 32;
#pragma unroll
  for (int j = 0; j < 32; j += 8) {
    const int k = kb + threadIdx.y + j, nn = nb + threadIdx.x;
    tile[threadIdx.y + j][threadIdx.x] = (k < K && nn < N) ? in[(size_t)k * N + nn] : 0.f;
  }
  __syncthreads();
#pragma unroll
  for (int j = 0; j < 32; j += 8) {
    const int nn = nb + threadIdx.y + j, k = kb + threadIdx.x;
    if (nn < N && k < K) out[(size_t)nn * K + k] = f2bf(tile[threadIdx.x][threadIdx.y + j]);
  }
}

__global__ void k_alpha(const float* __restrict__ tau, float* __restrict__ alpha, int n) {
  const int i = blockIdx.x * blockDim.x + threadIdx.x;
  if (i < n) {
    const float x = tau[i];
    const float sp = (x > 20.f) ? x : log1pf(expf(x));  // softplus
    alpha[i] = expf(-sp);
  }
}

// ---------------- persistent recurrent scan ----------------

__global__ void __launch_bounds__(256)
snn_scan_kernel(const u16* __restrict__ xpad,
                const u16* __restrict__ wt_in0, const u16* __restrict__ wt_out0,
                const u16* __restrict__ wt_in1, const u16* __restrict__ wt_out1,
                const float* __restrict__ b_in0, const float* __restrict__ b_out0,
                const float* __restrict__ b_in1, const float* __restrict__ b_out1,
                const float* __restrict__ thr0, const float* __restrict__ thr1,
                const float* __restrict__ alpha0, const float* __restrict__ alpha1,
                float* __restrict__ V0f, float* __restrict__ Df0,
                float* __restrict__ V1f, float* __restrict__ Df1,
                u16* __restrict__ cat0, u16* __restrict__ cat1,
                u16* __restrict__ out0, u16* __restrict__ seq,
                unsigned* __restrict__ bar) {
  __shared__ float part[NW][16][16];
  const int w = threadIdx.x >> 5;
  const int colb = blockIdx.x * 16;
  const int mm = threadIdx.x >> 4;       // padded batch row 0..15
  const int nn = threadIdx.x & 15;
  const int col = colb + nn;
  unsigned epoch = 0;

  const float a0c = alpha0[col], a1c = alpha1[col];
  const float bi0 = b_in0[col], bo0 = b_out0[col];
  const float bi1 = b_in1[col], bo1 = b_out1[col];
  const float th0 = thr0[col],  th1 = thr1[col];
  const bool live = (mm < Bc);   // pad rows stay zero (memset) and are never rewritten

  for (int t = 0; t < Sc; ++t) {
    // ---- A: Vt0 = alpha0*Vp0 + x_t @ w_in0 + b_in0 ; spike ; mirror flip ----
    {
      v8f acc = {};
      const int kc = Ic / NW;
      acc = wave_gemm_bf16(xpad + (size_t)t * Ic, (size_t)Sc * Ic,
                           wt_in0, Ic, colb, w * kc, (w + 1) * kc, acc);
      const float sum = block_reduce16(part, acc);
      if (live) {
        const float Vp = V0f[mm * Hc + col];
        const float Dp = Df0[mm * Hc + col];
        const float Vt = a0c * Vp + sum + bi0;
        const float s  = (Vt - th0) > 0.f ? 1.f : 0.f;
        const float Dt = Dp * (1.f - s) + (1.f - Dp) * s;
        V0f[mm * Hc + col] = Vt;
        Df0[mm * Hc + col] = Dt;
        cat0[mm * 2 * Hc + col]      = f2bf(Vt);
        cat0[mm * 2 * Hc + Hc + col] = f2bf(Dt);
      }
      grid_sync(bar, epoch);
    }
    // ---- B: out0 = tanh([Vt0,Dt0] @ w_out0 + b_out0) ----
    {
      v8f acc = {};
      const int kc = 2 * Hc / NW;
      acc = wave_gemm_bf16(cat0, 2 * Hc, wt_out0, 2 * Hc, colb,
                           w * kc, (w + 1) * kc, acc);
      const float sum = block_reduce16(part, acc);
      if (live) out0[mm * Hc + col] = f2bf(tanhf(sum + bo0));
      grid_sync(bar, epoch);
    }
    // ---- C: layer-1 membrane + spike + mirror ----
    {
      v8f acc = {};
      const int kc = Hc / NW;
      acc = wave_gemm_bf16(out0, Hc, wt_in1, Hc, colb, w * kc, (w + 1) * kc, acc);
      const float sum = block_reduce16(part, acc);
      if (live) {
        const float Vp = V1f[mm * Hc + col];
        const float Dp = Df1[mm * Hc + col];
        const float Vt = a1c * Vp + sum + bi1;
        const float s  = (Vt - th1) > 0.f ? 1.f : 0.f;
        const float Dt = Dp * (1.f - s) + (1.f - Dp) * s;
        V1f[mm * Hc + col] = Vt;
        Df1[mm * Hc + col] = Dt;
        cat1[mm * 2 * Hc + col]      = f2bf(Vt);
        cat1[mm * 2 * Hc + Hc + col] = f2bf(Dt);
      }
      grid_sync(bar, epoch);
    }
    // ---- D: out1 = tanh([Vt1,Dt1] @ w_out1 + b_out1) -> seq (overlaps next A) ----
    {
      v8f acc = {};
      const int kc = 2 * Hc / NW;
      acc = wave_gemm_bf16(cat1, 2 * Hc, wt_out1, 2 * Hc, colb,
                           w * kc, (w + 1) * kc, acc);
      const float sum = block_reduce16(part, acc);
      if (live) seq[((size_t)mm * Sc + t) * Hc + col] = f2bf(tanhf(sum + bo1));
      __syncthreads();   // protect LDS 'part' reuse; next grid barrier covers cross-block deps
    }
  }
}

// ---------------- LM head GEMM: [4096 x 1024] bf16 x [1024 x 32000] -> f32 ----------------

struct FragSet { Frag a0, a1, b0, b1; };

__device__ inline FragSet load_frags(const u16* __restrict__ a0p, const u16* __restrict__ a1p,
                                     const u16* __restrict__ b0p, const u16* __restrict__ b1p,
                                     int k) {
  FragSet f;
  f.a0.q[0] = *(const u32x4*)(a0p + k);  f.a0.q[1] = *(const u32x4*)(a0p + k + 16);
  f.a1.q[0] = *(const u32x4*)(a1p + k);  f.a1.q[1] = *(const u32x4*)(a1p + k + 16);
  f.b0.q[0] = *(const u32x4*)(b0p + k);  f.b0.q[1] = *(const u32x4*)(b0p + k + 16);
  f.b1.q[0] = *(const u32x4*)(b1p + k);  f.b1.q[1] = *(const u32x4*)(b1p + k + 16);
  return f;
}

__global__ void __launch_bounds__(256)
head_gemm_kernel(const u16* __restrict__ seqA, const u16* __restrict__ wt_head,
                 const float* __restrict__ b_head, float* __restrict__ out) {
  const int lane = threadIdx.x & 31;
  const int w  = threadIdx.x >> 5;
  const int hi = lane >> 4;
  const int l15 = lane & 15;
  const int mbase = blockIdx.y * 64  + (w & 1) * 32;   // 2x4 wave grid -> 64x128 block tile
  const int nbase = blockIdx.x * 128 + (w >> 1) * 32;

  const u16* a0p = seqA + (size_t)(mbase + l15) * Hc + hi * 8;
  const u16* a1p = a0p + (size_t)16 * Hc;
  const u16* b0p = wt_head + (size_t)(nbase + l15) * Hc + hi * 8;
  const u16* b1p = b0p + (size_t)16 * Hc;

  v8f acc00 = {}, acc01 = {}, acc10 = {}, acc11 = {};

  // Software-pipelined K loop (fully unrolled, K = 1024): issue next fragment loads
  // before consuming the current ones so WMMAs overlap outstanding global loads.
  FragSet cur = load_frags(a0p, a1p, b0p, b1p, 0);
#pragma unroll
  for (int k = 0; k < Hc; k += 32) {
    FragSet nxt;
    if (k + 32 < Hc) nxt = load_frags(a0p, a1p, b0p, b1p, k + 32);
    acc00 = __builtin_amdgcn_wmma_f32_16x16x32_bf16(false, cur.a0.v, false, cur.b0.v, (short)0, acc00, false, false);
    acc01 = __builtin_amdgcn_wmma_f32_16x16x32_bf16(false, cur.a0.v, false, cur.b1.v, (short)0, acc01, false, false);
    acc10 = __builtin_amdgcn_wmma_f32_16x16x32_bf16(false, cur.a1.v, false, cur.b0.v, (short)0, acc10, false, false);
    acc11 = __builtin_amdgcn_wmma_f32_16x16x32_bf16(false, cur.a1.v, false, cur.b1.v, (short)0, acc11, false, false);
    if (k + 32 < Hc) cur = nxt;
  }

  const int n0 = nbase + l15, n1 = n0 + 16;
  const float bb0 = b_head[n0], bb1 = b_head[n1];
  // 512 MB write-once logits: stream past L2 with non-temporal stores so w_head/seq stay resident.
#pragma unroll
  for (int r = 0; r < 8; ++r) {
    const size_t row0 = (size_t)(mbase + r + 8 * hi);
    const size_t row1 = row0 + 16;
    __builtin_nontemporal_store(acc00[r] + bb0, &out[row0 * Vc + n0]);
    __builtin_nontemporal_store(acc01[r] + bb1, &out[row0 * Vc + n1]);
    __builtin_nontemporal_store(acc10[r] + bb0, &out[row1 * Vc + n0]);
    __builtin_nontemporal_store(acc11[r] + bb1, &out[row1 * Vc + n1]);
  }
}

// ---------------- launch ----------------

extern "C" void kernel_launch(void* const* d_in, const int* in_sizes, int n_in,
                              void* d_out, int out_size, void* d_ws, size_t ws_size,
                              hipStream_t stream) {
  (void)in_sizes; (void)n_in; (void)out_size; (void)ws_size;

  const float* x       = (const float*)d_in[0];
  const float* w_in_0  = (const float*)d_in[1];
  const float* b_in_0  = (const float*)d_in[2];
  const float* tau_0   = (const float*)d_in[3];
  const float* thr_0   = (const float*)d_in[4];
  const float* w_out_0 = (const float*)d_in[5];
  const float* b_out_0 = (const float*)d_in[6];
  const float* w_in_1  = (const float*)d_in[7];
  const float* b_in_1  = (const float*)d_in[8];
  const float* tau_1   = (const float*)d_in[9];
  const float* thr_1   = (const float*)d_in[10];
  const float* w_out_1 = (const float*)d_in[11];
  const float* b_out_1 = (const float*)d_in[12];
  const float* w_head  = (const float*)d_in[13];
  const float* b_head  = (const float*)d_in[14];

  char* p = (char*)d_ws;
  auto carve = [&](size_t bytes) { char* r = p; p += (bytes + 255) & ~(size_t)255; return r; };

  // --- state region (zeroed every call; pad rows rely on these zeros) ---
  unsigned* bar = (unsigned*)carve(256);
  float* V0f = (float*)carve((size_t)16 * Hc * 4);
  float* Df0 = (float*)carve((size_t)16 * Hc * 4);
  float* V1f = (float*)carve((size_t)16 * Hc * 4);
  float* Df1 = (float*)carve((size_t)16 * Hc * 4);
  u16* cat0  = (u16*)carve((size_t)16 * 2 * Hc * 2);
  u16* cat1  = (u16*)carve((size_t)16 * 2 * Hc * 2);
  u16* out0  = (u16*)carve((size_t)16 * Hc * 2);
  const size_t state_bytes = (size_t)(p - (char*)d_ws);

  // --- derived constants / converted operands ---
  float* alpha0 = (float*)carve((size_t)Hc * 4);
  float* alpha1 = (float*)carve((size_t)Hc * 4);
  u16* xpad     = (u16*)carve((size_t)16 * Sc * Ic * 2);   // 16 rows; rows 4..15 zero
  u16* wt_in0   = (u16*)carve((size_t)Hc * Ic * 2);
  u16* wt_out0  = (u16*)carve((size_t)Hc * 2 * Hc * 2);
  u16* wt_in1   = (u16*)carve((size_t)Hc * Hc * 2);
  u16* wt_out1  = (u16*)carve((size_t)Hc * 2 * Hc * 2);
  u16* wt_head  = (u16*)carve((size_t)Vc * Hc * 2);
  u16* seq      = (u16*)carve((size_t)Bc * Sc * Hc * 2);

  hipMemsetAsync(d_ws, 0, state_bytes, stream);
  hipMemsetAsync(xpad, 0, (size_t)16 * Sc * Ic * 2, stream);

  // rows 0..3 of xpad == x converted to bf16 (identical linear layout)
  k_cvt_bf16<<<2048, 256, 0, stream>>>(x, xpad, (size_t)Bc * Sc * Ic);

  const dim3 tb(32, 8);
  k_transpose_bf16<<<dim3(Hc / 32, Ic / 32),     tb, 0, stream>>>(w_in_0,  wt_in0,  Ic,     Hc);
  k_transpose_bf16<<<dim3(Hc / 32, 2 * Hc / 32), tb, 0, stream>>>(w_out_0, wt_out0, 2 * Hc, Hc);
  k_transpose_bf16<<<dim3(Hc / 32, Hc / 32),     tb, 0, stream>>>(w_in_1,  wt_in1,  Hc,     Hc);
  k_transpose_bf16<<<dim3(Hc / 32, 2 * Hc / 32), tb, 0, stream>>>(w_out_1, wt_out1, 2 * Hc, Hc);
  k_transpose_bf16<<<dim3(Vc / 32, Hc / 32),     tb, 0, stream>>>(w_head,  wt_head, Hc,     Vc);

  k_alpha<<<Hc / 256, 256, 0, stream>>>(tau_0, alpha0, Hc);
  k_alpha<<<Hc / 256, 256, 0, stream>>>(tau_1, alpha1, Hc);

  snn_scan_kernel<<<NBLK, 256, 0, stream>>>(
      xpad, wt_in0, wt_out0, wt_in1, wt_out1,
      b_in_0, b_out_0, b_in_1, b_out_1, thr_0, thr_1, alpha0, alpha1,
      V0f, Df0, V1f, Df1, cat0, cat1, out0, seq, bar);

  head_gemm_kernel<<<dim3(Vc / 128, (Bc * Sc) / 64), 256, 0, stream>>>(
      seq, wt_head, b_head, (float*)d_out);
}
